// Attention_81028853007052
// MI455X (gfx1250) — compile-verified, matrix operand order checked by reading, not run on hardware
//
#include <hip/hip_runtime.h>

// ---------------------------------------------------------------------------
// Problem constants (from reference)
// ---------------------------------------------------------------------------
#define HEADS    16
#define DIM_HEAD 64
#define DIM      1024
#define INNER    1024
#define BATCH    2
#define SEQ      2048
#define NKEY     2050            // 2 null kv + 2048
#define JPAD     2080            // padded to multiple of 32
#define MTOK     (BATCH * SEQ)   // 4096 tokens
#define SCALE_F  8.0f
#define NEG_BIG  (-3.0e38f)

// ---------------------------------------------------------------------------
// gfx1250 async global->LDS path (guarded; falls back to uint4 copies)
// ---------------------------------------------------------------------------
#if defined(__HIP_DEVICE_COMPILE__) &&                                   \
    __has_builtin(__builtin_amdgcn_global_load_async_to_lds_b128) &&     \
    __has_builtin(__builtin_amdgcn_s_wait_asynccnt)
#define HAVE_ASYNC_LDS 1
#else
#define HAVE_ASYNC_LDS 0
#endif

#define AS1 __attribute__((address_space(1)))
#define AS3 __attribute__((address_space(3)))

typedef int v4i_vec __attribute__((vector_size(16)));   // matches builtin proto

__device__ __forceinline__ void copy16_g2l(void* lds_dst, const void* gsrc) {
#if HAVE_ASYNC_LDS
  // generic->global pointer value is identity; generic LDS pointer's low 32
  // bits are the LDS offset, so truncate for the addrspace(3) pointer.
  __builtin_amdgcn_global_load_async_to_lds_b128(
      (AS1 v4i_vec*)(unsigned long long)gsrc,
      (AS3 v4i_vec*)(unsigned int)(unsigned long long)lds_dst,
      /*imm offset=*/0, /*cpol=*/0);
#else
  *(uint4*)lds_dst = *(const uint4*)gsrc;
#endif
}
__device__ __forceinline__ void async_drain() {
#if HAVE_ASYNC_LDS
  __builtin_amdgcn_s_wait_asynccnt(0);
#endif
}

// ---------------------------------------------------------------------------
// WMMA fragment plumbing (gfx1250, wave32)
// ---------------------------------------------------------------------------
typedef __attribute__((ext_vector_type(16))) __bf16 v16bf;
typedef __attribute__((ext_vector_type(8)))  float  v8f;

union Frag16 { unsigned int u[8]; v16bf v; };

__device__ __forceinline__ unsigned short f2bf(float f) {
  unsigned int u = __builtin_bit_cast(unsigned int, f);
  u += 0x7FFFu + ((u >> 16) & 1u);          // round to nearest even
  return (unsigned short)(u >> 16);
}

// K index for dword v of a 16-bit A/B fragment (lane half = lane>>4).
__device__ __forceinline__ int kmap(int v, int half) {
  return ((v < 4) ? (v * 2) : (16 + (v - 4) * 2)) + 8 * half;
}

__device__ __forceinline__ v8f wmma_bf16(v16bf a, v16bf b, v8f c) {
  return __builtin_amdgcn_wmma_f32_16x16x32_bf16(
      /*neg_a=*/false, a, /*neg_b=*/false, b,
      /*c_mod=*/(short)0, c, /*reuse_a=*/false, /*reuse_b=*/false);
}

// ---------------------------------------------------------------------------
// fp32 -> bf16 pack
// ---------------------------------------------------------------------------
__global__ __launch_bounds__(256) void pack_bf16(const float* __restrict__ in,
                                                 unsigned short* __restrict__ out,
                                                 int n) {
  int i = blockIdx.x * 256 + threadIdx.x;
  if (i < n) out[i] = f2bf(in[i]);
}

// ---------------------------------------------------------------------------
// LayerNorm(x, gamma) -> bf16, one block per token (dim = 1024)
// ---------------------------------------------------------------------------
__global__ __launch_bounds__(256) void ln_pack(const float* __restrict__ x,
                                               const float* __restrict__ gamma,
                                               unsigned short* __restrict__ out) {
  __shared__ float red[256];
  const int tid = threadIdx.x;
  const float* xr = x + (size_t)blockIdx.x * DIM;
  float vals[4];
  float s = 0.f;
#pragma unroll
  for (int i = 0; i < 4; ++i) { vals[i] = xr[tid + i * 256]; s += vals[i]; }
  red[tid] = s; __syncthreads();
  for (int st = 128; st > 0; st >>= 1) {
    if (tid < st) red[tid] += red[tid + st];
    __syncthreads();
  }
  const float mu = red[0] * (1.f / DIM);
  __syncthreads();
  float vs = 0.f;
#pragma unroll
  for (int i = 0; i < 4; ++i) { float d = vals[i] - mu; vs += d * d; }
  red[tid] = vs; __syncthreads();
  for (int st = 128; st > 0; st >>= 1) {
    if (tid < st) red[tid] += red[tid + st];
    __syncthreads();
  }
  const float inv = rsqrtf(red[0] * (1.f / DIM) + 1e-5f);
  unsigned short* orow = out + (size_t)blockIdx.x * DIM;
#pragma unroll
  for (int i = 0; i < 4; ++i) {
    const int c = tid + i * 256;
    orow[c] = f2bf((vals[i] - mu) * inv * gamma[c]);
  }
}

// ---------------------------------------------------------------------------
// bf16 x bf16 -> f32 GEMM, 128x128 block tile, K step 32, LDS-staged (async).
// 8 waves arranged 4(M) x 2(N); wave computes 32x64 = 2x4 WMMA accumulators.
// M, N multiples of 128; K multiple of 32.
// ---------------------------------------------------------------------------
#define BK 32
__global__ __launch_bounds__(256) void gemm_bf16(const unsigned short* __restrict__ A,
                                                 const unsigned short* __restrict__ B,
                                                 float* __restrict__ C,
                                                 int M, int N, int K) {
  __shared__ unsigned short lA[128 * BK];
  __shared__ unsigned short lB[BK * 128];
  const int tid  = threadIdx.x;
  const int lane = tid & 31, w = tid >> 5;
  const int wm = w & 3, wn = w >> 2;
  const int half = lane >> 4, l16 = lane & 15;
  const int rowBlk = blockIdx.x * 128;
  const int colBlk = blockIdx.y * 128;

  v8f acc[2][4];
#pragma unroll
  for (int i = 0; i < 2; ++i)
#pragma unroll
    for (int j = 0; j < 4; ++j)
#pragma unroll
      for (int e = 0; e < 8; ++e) acc[i][j][e] = 0.f;

  for (int k0 = 0; k0 < K; k0 += BK) {
    // cooperative tile staging, 16B (8 bf16) per transaction, async to LDS
#pragma unroll
    for (int it = 0; it < 2; ++it) {
      const int idx = tid + it * 256;                  // 0..511
      const int ar = idx >> 2, ac = (idx & 3) * 8;     // 128 x 32
      copy16_g2l(&lA[ar * BK + ac],
                 &A[(size_t)(rowBlk + ar) * K + k0 + ac]);
      const int br = idx >> 4, bc = (idx & 15) * 8;    // 32 x 128
      copy16_g2l(&lB[br * 128 + bc],
                 &B[(size_t)(k0 + br) * N + colBlk + bc]);
    }
    async_drain();
    __syncthreads();

    Frag16 af[2], bf[4];
#pragma unroll
    for (int fm = 0; fm < 2; ++fm) {
      const int row = wm * 32 + fm * 16 + l16;
#pragma unroll
      for (int v = 0; v < 8; ++v)
        af[fm].u[v] = *(const unsigned int*)&lA[row * BK + kmap(v, half)];
    }
#pragma unroll
    for (int fn = 0; fn < 4; ++fn) {
      const int col = wn * 64 + fn * 16 + l16;
#pragma unroll
      for (int v = 0; v < 8; ++v) {
        const int kk = kmap(v, half);
        const unsigned lo = lB[kk * 128 + col];
        const unsigned hi = lB[(kk + 1) * 128 + col];
        bf[fn].u[v] = lo | (hi << 16);
      }
    }
#pragma unroll
    for (int fm = 0; fm < 2; ++fm)
#pragma unroll
      for (int fn = 0; fn < 4; ++fn)
        acc[fm][fn] = wmma_bf16(af[fm].v, bf[fn].v, acc[fm][fn]);
    __syncthreads();
  }

  // C layout: row = v + 8*(lane>>4), col = lane&15
#pragma unroll
  for (int fm = 0; fm < 2; ++fm)
#pragma unroll
    for (int fn = 0; fn < 4; ++fn)
#pragma unroll
      for (int v = 0; v < 8; ++v) {
        const int row = rowBlk + wm * 32 + fm * 16 + v + 8 * half;
        const int col = colBlk + wn * 64 + fn * 16 + l16;
        C[(size_t)row * N + col] = acc[fm][fn][v];
      }
}

// ---------------------------------------------------------------------------
// q: l2-normalize per (b,m,h), * q_scale, -> bf16 in (b,h,n,Dh) layout
// ---------------------------------------------------------------------------
__global__ __launch_bounds__(256) void prep_q(const float* __restrict__ qf,
                                              const float* __restrict__ q_scale,
                                              unsigned short* __restrict__ qb) {
  const int idx = blockIdx.x * 256 + threadIdx.x;  // over b*n*H
  if (idx >= BATCH * SEQ * HEADS) return;
  const int hh  = idx & (HEADS - 1);
  const int tok = idx >> 4;                        // b*SEQ + m
  const int bb  = tok / SEQ;
  const int m   = tok - bb * SEQ;
  const float* src = qf + (size_t)tok * INNER + hh * DIM_HEAD;
  float ss = 0.f;
  for (int d = 0; d < DIM_HEAD; ++d) { const float t = src[d]; ss += t * t; }
  const float inv = 1.f / fmaxf(sqrtf(ss), 1e-12f);
  unsigned short* dst = qb + (((size_t)(bb * HEADS + hh)) * SEQ + m) * DIM_HEAD;
  for (int d = 0; d < DIM_HEAD; ++d) dst[d] = f2bf(src[d] * inv * q_scale[d]);
}

// ---------------------------------------------------------------------------
// k/v: prepend null kv, l2-normalize k * k_scale, pad keys to JPAD.
// k row-major (b, JPAD, Dh); v TRANSPOSED (b, Dh, JPAD) so both attention
// B-fragments read contiguous dwords.
// ---------------------------------------------------------------------------
__global__ __launch_bounds__(256) void prep_kv(const float* __restrict__ kvf,
                                               const float* __restrict__ k_scale,
                                               const float* __restrict__ null_kv,
                                               unsigned short* __restrict__ kb,
                                               unsigned short* __restrict__ vtb) {
  const int idx = blockIdx.x * 256 + threadIdx.x;  // over b * JPAD
  if (idx >= BATCH * JPAD) return;
  const int j  = idx % JPAD;
  const int bb = idx / JPAD;
  float kv2[DIM_HEAD], vv[DIM_HEAD];
  if (j < 2) {
    for (int d = 0; d < DIM_HEAD; ++d) {
      kv2[d] = null_kv[(0 * 2 + j) * DIM_HEAD + d];
      vv[d]  = null_kv[(1 * 2 + j) * DIM_HEAD + d];
    }
  } else if (j < NKEY) {
    const float* r = kvf + ((size_t)bb * SEQ + (j - 2)) * (2 * DIM_HEAD);
    for (int d = 0; d < DIM_HEAD; ++d) { kv2[d] = r[d]; vv[d] = r[DIM_HEAD + d]; }
  } else {
    for (int d = 0; d < DIM_HEAD; ++d) { kv2[d] = 0.f; vv[d] = 0.f; }
  }
  float ss = 0.f;
  for (int d = 0; d < DIM_HEAD; ++d) ss += kv2[d] * kv2[d];
  const float inv = (j < NKEY) ? 1.f / fmaxf(sqrtf(ss), 1e-12f) : 0.f;
  unsigned short* kr = kb + ((size_t)bb * JPAD + j) * DIM_HEAD;
  for (int d = 0; d < DIM_HEAD; ++d) kr[d] = f2bf(kv2[d] * inv * k_scale[d]);
  for (int d = 0; d < DIM_HEAD; ++d)
    vtb[((size_t)bb * DIM_HEAD + d) * JPAD + j] = f2bf(vv[d]);
}

// ---------------------------------------------------------------------------
// Flash attention: block = 128 query rows of one (b,h); 8 waves x 16 rows.
// Block-uniform 32-key tiles staged ONCE per block into LDS (async path);
// extra tiles for early waves are fully masked and contribute exactly 0.
// S and PV via v_wmma_f32_16x16x32_bf16; online softmax in f32 with 16-lane
// shfl reductions; P redistributed C-layout -> A-layout through per-wave LDS.
// ---------------------------------------------------------------------------
__global__ __launch_bounds__(256) void flash_attn(const unsigned short* __restrict__ qb,
                                                  const unsigned short* __restrict__ kb,
                                                  const unsigned short* __restrict__ vtb,
                                                  unsigned short* __restrict__ ao) {
  __shared__ unsigned short lK[32 * 64];            // K tile, (j, d) row-major
  __shared__ unsigned short lV[64 * 32];            // V^T tile, (d, j)
  __shared__ unsigned short lP[8][512];             // per-wave 16x32 bf16 P tile
  const int tid  = threadIdx.x;
  const int lane = tid & 31, w = tid >> 5;
  const int half = lane >> 4, l16 = lane & 15;
  const int bh = blockIdx.y;                        // b*HEADS + h
  const int bb = bh >> 4, h = bh & 15;
  const int blkrow  = blockIdx.x * 128;
  const int rowbase = blkrow + w * 16;

  // Persistent Q fragments: 16 rows x 64 dims -> two 16x32 A-frags
  Frag16 qa[2];
  {
    const unsigned short* qr =
        qb + (((size_t)bh * SEQ) + rowbase + l16) * DIM_HEAD;
#pragma unroll
    for (int c = 0; c < 2; ++c)
#pragma unroll
      for (int v = 0; v < 8; ++v)
        qa[c].u[v] = *(const unsigned int*)&qr[c * 32 + kmap(v, half)];
  }

  v8f o[4];
#pragma unroll
  for (int nf = 0; nf < 4; ++nf)
#pragma unroll
    for (int e = 0; e < 8; ++e) o[nf][e] = 0.f;
  float mrun[8], lrun[8];
#pragma unroll
  for (int v = 0; v < 8; ++v) { mrun[v] = NEG_BIG; lrun[v] = 0.f; }

  const unsigned short* kbase = kb + (size_t)bb * JPAD * DIM_HEAD;
  const unsigned short* vbase = vtb + (size_t)bb * DIM_HEAD * JPAD;

  // Block-uniform causal horizon: max row = blkrow+127, cols <= row+2;
  // covers the 64-row non-causal prefix window (cols < 66) as well.
  const int jend   = blkrow + 130;
  const int jtiles = (jend + 31) >> 5;

  // per-thread staging coordinates (K: 32x64, V^T: 64x32, 16B each)
  const int kr = tid >> 3, kc = (tid & 7) * 8;
  const int vr = tid >> 2, vc = (tid & 3) * 8;

  for (int jt = 0; jt < jtiles; ++jt) {
    const int j0 = jt * 32;
    // stage K/V tiles once per block (async global->LDS when available)
    copy16_g2l(&lK[kr * 64 + kc], kbase + (size_t)(j0 + kr) * DIM_HEAD + kc);
    copy16_g2l(&lV[vr * 32 + vc], vbase + (size_t)vr * JPAD + j0 + vc);
    if (jt + 1 < jtiles)
      __builtin_prefetch(kbase + (size_t)(j0 + 32) * DIM_HEAD, 0, 1);
    async_drain();
    __syncthreads();

    // S = Q K^T for 32 keys: two 16x16 C-frags, each accumulating 2 K-steps
    v8f s[2];
#pragma unroll
    for (int jh = 0; jh < 2; ++jh)
#pragma unroll
      for (int e = 0; e < 8; ++e) s[jh][e] = 0.f;
#pragma unroll
    for (int jh = 0; jh < 2; ++jh) {
      const unsigned short* krow = &lK[(jh * 16 + l16) * 64];
#pragma unroll
      for (int c = 0; c < 2; ++c) {
        Frag16 bkf;
#pragma unroll
        for (int v = 0; v < 8; ++v)
          bkf.u[v] = *(const unsigned int*)&krow[c * 32 + kmap(v, half)];
        s[jh] = wmma_bf16(qa[c].v, bkf.v, s[jh]);
      }
    }

    // Mask + online softmax; each row lives across one 16-lane half.
    float pm[2][8];
#pragma unroll
    for (int v = 0; v < 8; ++v) {
      const int row = rowbase + v + 8 * half;
      float mx = NEG_BIG;
#pragma unroll
      for (int jh = 0; jh < 2; ++jh) {
        const int col = j0 + jh * 16 + l16;
        float sv = s[jh][v] * SCALE_F;
        const bool masked =
            (col >= NKEY) || ((col > row + 2) && !((row < 64) && (col < 66)));
        sv = masked ? NEG_BIG : sv;
        pm[jh][v] = sv;
        mx = fmaxf(mx, sv);
      }
#pragma unroll
      for (int d = 1; d < 16; d <<= 1) mx = fmaxf(mx, __shfl_xor(mx, d, 32));
      const float nm   = fmaxf(mrun[v], mx);
      const float corr = __expf(mrun[v] - nm);
      float rs = 0.f;
#pragma unroll
      for (int jh = 0; jh < 2; ++jh) {
        const float p = __expf(pm[jh][v] - nm);
        pm[jh][v] = p;
        rs += p;
      }
#pragma unroll
      for (int d = 1; d < 16; d <<= 1) rs += __shfl_xor(rs, d, 32);
      mrun[v] = nm;
      lrun[v] = lrun[v] * corr + rs;
#pragma unroll
      for (int nf = 0; nf < 4; ++nf) o[nf][v] *= corr;
    }

    // Redistribute P (C-layout) -> A-layout through this wave's LDS slice.
    unsigned short* myP = &lP[w][0];
#pragma unroll
    for (int v = 0; v < 8; ++v)
#pragma unroll
      for (int jh = 0; jh < 2; ++jh)
        myP[(v + 8 * half) * 32 + jh * 16 + l16] = f2bf(pm[jh][v]);
    __builtin_amdgcn_wave_barrier();   // same-wave DS ops retire in order
    Frag16 pa;
#pragma unroll
    for (int v = 0; v < 8; ++v)
      pa.u[v] = *(const unsigned int*)&myP[l16 * 32 + kmap(v, half)];

    // O += P V : V^T tile rows give contiguous (j, j+1) dword pairs.
#pragma unroll
    for (int nf = 0; nf < 4; ++nf) {
      Frag16 bvf;
      const unsigned short* vrow = &lV[(nf * 16 + l16) * 32];
#pragma unroll
      for (int v = 0; v < 8; ++v)
        bvf.u[v] = *(const unsigned int*)&vrow[kmap(v, half)];
      o[nf] = wmma_bf16(pa.v, bvf.v, o[nf]);
    }
    __syncthreads();                   // protect lK/lV before next staging
  }

  // Normalize and emit bf16 attention output, token-major with head columns.
#pragma unroll
  for (int v = 0; v < 8; ++v) {
    const float invl = 1.f / fmaxf(lrun[v], 1e-20f);
    const int row = rowbase + v + 8 * half;
    unsigned short* orow =
        ao + ((size_t)bb * SEQ + row) * INNER + h * DIM_HEAD;
#pragma unroll
    for (int nf = 0; nf < 4; ++nf)
      orow[nf * 16 + l16] = f2bf(o[nf][v] * invl);
  }
}

// ---------------------------------------------------------------------------
// Host-side launch sequence
// ---------------------------------------------------------------------------
extern "C" void kernel_launch(void* const* d_in, const int* in_sizes, int n_in,
                              void* d_out, int out_size, void* d_ws, size_t ws_size,
                              hipStream_t stream) {
  const float* x       = (const float*)d_in[0];
  const float* gamma   = (const float*)d_in[1];
  const float* Wq      = (const float*)d_in[2];
  const float* Wkv     = (const float*)d_in[3];
  const float* q_scale = (const float*)d_in[4];
  const float* k_scale = (const float*)d_in[5];
  const float* null_kv = (const float*)d_in[6];
  const float* Wout    = (const float*)d_in[7];
  float* out = (float*)d_out;

  char* ws = (char*)d_ws;
  size_t off = 0;
  auto take = [&](size_t bytes) -> char* {
    char* p = ws + off;
    off = (off + bytes + 255) & ~(size_t)255;
    return p;
  };
  unsigned short* xb    = (unsigned short*)take((size_t)MTOK * DIM * 2);
  unsigned short* xnb   = (unsigned short*)take((size_t)MTOK * DIM * 2);
  unsigned short* wqb   = (unsigned short*)take((size_t)DIM * INNER * 2);
  unsigned short* wkvb  = (unsigned short*)take((size_t)DIM * 2 * DIM_HEAD * 2);
  unsigned short* woutb = (unsigned short*)take((size_t)INNER * DIM * 2);
  float*          qf    = (float*)take((size_t)MTOK * INNER * 4);
  float*          kvf   = (float*)take((size_t)MTOK * 2 * DIM_HEAD * 4);
  unsigned short* qbf   = (unsigned short*)take((size_t)MTOK * INNER * 2);
  unsigned short* kbb   = (unsigned short*)take((size_t)BATCH * JPAD * DIM_HEAD * 2);
  unsigned short* vtb   = (unsigned short*)take((size_t)BATCH * JPAD * DIM_HEAD * 2);
  unsigned short* aob   = (unsigned short*)take((size_t)MTOK * INNER * 2);

  pack_bf16<<<(MTOK * DIM + 255) / 256, 256, 0, stream>>>(x, xb, MTOK * DIM);
  pack_bf16<<<(DIM * INNER + 255) / 256, 256, 0, stream>>>(Wq, wqb, DIM * INNER);
  pack_bf16<<<(DIM * 2 * DIM_HEAD + 255) / 256, 256, 0, stream>>>(
      Wkv, wkvb, DIM * 2 * DIM_HEAD);
  pack_bf16<<<(INNER * DIM + 255) / 256, 256, 0, stream>>>(Wout, woutb, INNER * DIM);
  ln_pack<<<MTOK, 256, 0, stream>>>(x, gamma, xnb);

  gemm_bf16<<<dim3(MTOK / 128, INNER / 128), 256, 0, stream>>>(
      xnb, wqb, qf, MTOK, INNER, DIM);
  gemm_bf16<<<dim3(MTOK / 128, 1), 256, 0, stream>>>(
      xb, wkvb, kvf, MTOK, 2 * DIM_HEAD, DIM);

  prep_q<<<(BATCH * SEQ * HEADS + 255) / 256, 256, 0, stream>>>(qf, q_scale, qbf);
  prep_kv<<<(BATCH * JPAD + 255) / 256, 256, 0, stream>>>(
      kvf, k_scale, null_kv, kbb, vtb);

  flash_attn<<<dim3(SEQ / 128, BATCH * HEADS), 256, 0, stream>>>(qbf, kbb, vtb, aob);

  gemm_bf16<<<dim3(MTOK / 128, DIM / 128), 256, 0, stream>>>(
      aob, woutb, out, MTOK, DIM, INNER);
}
